// Qwen3_64312840290534
// MI455X (gfx1250) — matrix operand order repaired and checked by reading, not compiled, not run on hardware
//
#include <hip/hip_runtime.h>
#include <hip/hip_bf16.h>
#include <math.h>

// ---------------------------------------------------------------------------
// Qwen3-style forward pass for MI455X (gfx1250), bf16 WMMA everywhere.
// Round 3: hardware packed f32->bf16 conversion (v_cvt_pk_bf16_f32) when the
// builtin exists, manual RNE fallback otherwise. Rest unchanged from round 2.
// ---------------------------------------------------------------------------

typedef __attribute__((ext_vector_type(16))) __bf16 v16bf;
typedef __attribute__((ext_vector_type(2)))  __bf16 v2bf;
typedef __attribute__((ext_vector_type(8)))  float  v8f;

static constexpr int cV   = 32000;
static constexpr int cL   = 4;
static constexpr int cD   = 1024;
static constexpr int cFF  = 2736;
static constexpr int cH   = 16;
static constexpr int cHKV = 8;
static constexpr int cDH  = 128;
static constexpr int cT   = 2048;
static constexpr float cEPS = 1e-6f;

// ----- f32 -> bf16 conversion ----------------------------------------------

__device__ __forceinline__ unsigned pack_bf2(float lo, float hi) {
#if __has_builtin(__builtin_amdgcn_cvt_pk_bf16_f32)
  v2bf p = __builtin_amdgcn_cvt_pk_bf16_f32(lo, hi);
  return __builtin_bit_cast(unsigned, p);
#else
  unsigned ul = __builtin_bit_cast(unsigned, lo);
  unsigned uh = __builtin_bit_cast(unsigned, hi);
  ul += 0x7FFFu + ((ul >> 16) & 1u);        // round-to-nearest-even
  uh += 0x7FFFu + ((uh >> 16) & 1u);
  return (ul >> 16) | (uh & 0xFFFF0000u);
#endif
}

__device__ __forceinline__ unsigned short f2bf(float f) {
  return (unsigned short)(pack_bf2(f, 0.f) & 0xFFFFu);
}

__device__ __forceinline__ v8f wmma_bf16(v16bf a, v16bf b, v8f c) {
  return __builtin_amdgcn_wmma_f32_16x16x32_bf16(
      /*neg_a=*/false, a, /*neg_b=*/false, b,
      /*c_mod=*/(short)0, c, /*reuse_a=*/false, /*reuse_b=*/false);
}

// A-fragment (16x32, M x K) from LDS row-major [16][stride] bf16.
__device__ __forceinline__ v16bf frag_a(const unsigned short* p, int stride,
                                        int col, int lane) {
  int r = lane & 15, h = lane >> 4;
  union { v16bf v; unsigned u[8]; } f;
#pragma unroll
  for (int i = 0; i < 8; ++i) {
    int kb = 16 * h + 2 * i + col;
    f.u[i] = *(const unsigned*)(p + (size_t)r * stride + kb);
  }
  return f.v;
}

// B-fragment (32x16, K x N) from LDS [N][stride] (n-major) bf16.
__device__ __forceinline__ v16bf frag_b(const unsigned short* p, int stride,
                                        int col, int lane) {
  int r = lane & 15, h = lane >> 4;
  union { v16bf v; unsigned u[8]; } f;
#pragma unroll
  for (int i = 0; i < 8; ++i) {
    int kb = 16 * h + 2 * i + col;
    f.u[i] = *(const unsigned*)(p + (size_t)r * stride + kb);
  }
  return f.v;
}

// ----- embedding gather -----------------------------------------------------

__global__ __launch_bounds__(256) void embed_k(const int* __restrict__ ids,
                                               const float* __restrict__ emb,
                                               float* __restrict__ x) {
  int i = blockIdx.x * 256 + threadIdx.x;           // T*D elements
  int t = i >> 10, d = i & 1023;
  x[i] = emb[(size_t)ids[t] * cD + d];
}

// ----- token RMSNorm --------------------------------------------------------

__global__ __launch_bounds__(256) void rmsnorm_k(const float* __restrict__ x,
                                                 const float* __restrict__ w,
                                                 float* __restrict__ out) {
  int t = blockIdx.x;
  const float* xr = x + (size_t)t * cD;
  float* yr = out + (size_t)t * cD;
  float s = 0.f;
  for (int i = threadIdx.x; i < cD; i += 256) { float v = xr[i]; s += v * v; }
  __shared__ float red[256];
  red[threadIdx.x] = s;
  __syncthreads();
  for (int off = 128; off > 0; off >>= 1) {
    if ((int)threadIdx.x < off) red[threadIdx.x] += red[threadIdx.x + off];
    __syncthreads();
  }
  float r = rsqrtf(red[0] / (float)cD + cEPS);
  for (int i = threadIdx.x; i < cD; i += 256) yr[i] = xr[i] * r * w[i];
}

// ----- bf16 WMMA GEMM: C[M,N] = A[M,K] * W[N,K]^T (+res) --------------------

#define GBM 64
#define GBN 128
#define GBK 32
#define GLS 40   // padded LDS stride (bf16 elements)

__global__ __launch_bounds__(256) void gemm_bf16_k(
    const float* __restrict__ A, const float* __restrict__ W,
    const float* __restrict__ res, float* __restrict__ C,
    int M, int N, int K) {
  __shared__ unsigned short As[GBM * GLS];   // 5 KB
  __shared__ unsigned short Bs[GBN * GLS];   // 10 KB
  int tid = threadIdx.x, lane = tid & 31, wave = tid >> 5;
  int m0 = blockIdx.y * GBM, n0 = blockIdx.x * GBN;
  int wm = (wave & 3) * 16;           // wave's 16-row strip
  int wn = (wave >> 2) * 64;          // wave's four 16-col tiles
  v8f acc[4];
#pragma unroll
  for (int j = 0; j < 4; ++j) { v8f z = {}; acc[j] = z; }

  const bool mnfull = (m0 + GBM <= M) && (n0 + GBN <= N);
  // fast-path addressing (one row / contiguous K-segment per thread)
  const int ar = tid >> 2, aks = (tid & 3) * 8;     // A: 64 rows x 8 k
  const int br = tid >> 1, bks = (tid & 1) * 16;    // B: 128 rows x 16 k

  int nk = (K + GBK - 1) / GBK;
  for (int kc = 0; kc < nk; ++kc) {
    int k0 = kc * GBK;
    __syncthreads();
    if (mnfull && (k0 + GBK <= K)) {
      // ---- branch-free vector staging ----
      const float* sa = A + (size_t)(m0 + ar) * K + k0 + aks;
      float4 a0 = *(const float4*)sa;
      float4 a1 = *(const float4*)(sa + 4);
      uint4 pa = { pack_bf2(a0.x, a0.y), pack_bf2(a0.z, a0.w),
                   pack_bf2(a1.x, a1.y), pack_bf2(a1.z, a1.w) };
      *(uint4*)&As[ar * GLS + aks] = pa;

      const float* sb = W + (size_t)(n0 + br) * K + k0 + bks;
      float4 b0 = *(const float4*)sb;
      float4 b1 = *(const float4*)(sb + 4);
      float4 b2 = *(const float4*)(sb + 8);
      float4 b3 = *(const float4*)(sb + 12);
      uint4 pb0 = { pack_bf2(b0.x, b0.y), pack_bf2(b0.z, b0.w),
                    pack_bf2(b1.x, b1.y), pack_bf2(b1.z, b1.w) };
      uint4 pb1 = { pack_bf2(b2.x, b2.y), pack_bf2(b2.z, b2.w),
                    pack_bf2(b3.x, b3.y), pack_bf2(b3.z, b3.w) };
      *(uint4*)&Bs[br * GLS + bks] = pb0;
      *(uint4*)&Bs[br * GLS + bks + 8] = pb1;
    } else {
      // ---- guarded tail path ----
#pragma unroll
      for (int i = 0; i < 8; ++i) {
        int idx = tid + i * 256;                 // 2048 A elems
        int r = idx >> 5, kk = idx & 31;
        float a = 0.f;
        if (k0 + kk < K && m0 + r < M) a = A[(size_t)(m0 + r) * K + k0 + kk];
        As[r * GLS + kk] = f2bf(a);
      }
#pragma unroll
      for (int i = 0; i < 16; ++i) {
        int idx = tid + i * 256;                 // 4096 B elems
        int r = idx >> 5, kk = idx & 31;
        float b = 0.f;
        if (k0 + kk < K && n0 + r < N) b = W[(size_t)(n0 + r) * K + k0 + kk];
        Bs[r * GLS + kk] = f2bf(b);
      }
    }
    // prefetch next weight chunk (global_prefetch_b8)
    if (kc + 1 < nk) {
      int pr = n0 + br, pk = k0 + GBK + bks;
      if (pr < N && pk < K)
        __builtin_prefetch(&W[(size_t)pr * K + pk], 0, 1);
    }
    __syncthreads();
    v16bf af = frag_a(As + wm * GLS, GLS, 0, lane);
#pragma unroll
    for (int j = 0; j < 4; ++j) {
      v16bf bfj = frag_b(Bs + (wn + j * 16) * GLS, GLS, 0, lane);
      acc[j] = wmma_bf16(af, bfj, acc[j]);
    }
  }
  // epilogue: C layout — lane half h covers rows 8h..8h+7, col = lane&15
  int h = lane >> 4, nn = lane & 15;
#pragma unroll
  for (int j = 0; j < 4; ++j) {
    int n = n0 + wn + j * 16 + nn;
    if (n >= N) continue;
#pragma unroll
    for (int v = 0; v < 8; ++v) {
      int m = m0 + wm + h * 8 + v;
      if (m >= M) continue;
      float val = acc[j][v];
      if (res) val += res[(size_t)m * N + n];
      C[(size_t)m * N + n] = val;
    }
  }
}

// ----- per-head RMSNorm + RoPE (one wave per (token, head)) -----------------

__global__ __launch_bounds__(32) void rope_rms_k(float* __restrict__ buf,
                                                 int nheads, int rowstride,
                                                 const float* __restrict__ nw) {
  int b = blockIdx.x;
  int t = b / nheads, hh = b % nheads;
  float* p = buf + (size_t)t * rowstride + hh * cDH;
  int lane = threadIdx.x;
  float v[4];
  float s = 0.f;
#pragma unroll
  for (int j = 0; j < 4; ++j) { v[j] = p[lane + 32 * j]; s += v[j] * v[j]; }
#pragma unroll
  for (int off = 16; off > 0; off >>= 1) s += __shfl_xor(s, off, 32);
  float r = rsqrtf(s / (float)cDH + cEPS);
  float nv[4];
#pragma unroll
  for (int j = 0; j < 4; ++j) {
    int idx = lane + 32 * j;
    nv[j] = v[j] * r * nw[idx];
  }
  float out[4];
#pragma unroll
  for (int j = 0; j < 4; ++j) {
    int idx = lane + 32 * j;
    int i = idx & 63;
    float f = (float)t * __expf(-13.815510558f * ((float)(2 * i) / (float)cDH));
    float cs = cosf(f), sn = sinf(f);
    float rh = (j < 2) ? -nv[j + 2] : nv[j - 2];
    out[j] = nv[j] * cs + rh * sn;
  }
#pragma unroll
  for (int j = 0; j < 4; ++j) p[lane + 32 * j] = out[j];
}

// ----- flash attention: one wave per (head, 16-query tile) ------------------

#define KSTR 136   // padded bf16 stride for 128-wide tiles
#define VSTR 40    // padded bf16 stride for 32-wide tiles

__global__ __launch_bounds__(32) void attn_k(const float* __restrict__ q,
                                             const float* __restrict__ k,
                                             const float* __restrict__ v,
                                             float* __restrict__ ctx) {
  __shared__ unsigned short qs[16 * KSTR];   // Q tile   16 x 128 (m-major)
  __shared__ unsigned short ks[32 * KSTR];   // K block  32 x 128 (key-major)
  __shared__ unsigned short vs[128 * VSTR];  // V block  128 x 32 (feat-major)
  __shared__ unsigned short ps[16 * VSTR];   // P tile   16 x 32  (m-major)
  int lane = threadIdx.x;
  int hh = blockIdx.x;                        // head
  int q0 = blockIdx.y * 16;                   // first query row
  int kvh = hh / (cH / cHKV);
  const float scale = 0.088388347648318447f;  // 1/sqrt(128)

  // stage Q (f32 -> bf16), vectorized: 512 float4 chunks
#pragma unroll
  for (int it = 0; it < 16; ++it) {
    int idx = lane + it * 32;
    int r = idx >> 5, c4 = (idx & 31) * 4;
    float4 qv = *(const float4*)(q + (size_t)(q0 + r) * (cH * cDH) + hh * cDH + c4);
    uint2 pk = { pack_bf2(qv.x, qv.y), pack_bf2(qv.z, qv.w) };
    *(uint2*)&qs[r * KSTR + c4] = pk;
  }
  __syncthreads();
  v16bf qa[4];
#pragma unroll
  for (int kc = 0; kc < 4; ++kc) qa[kc] = frag_a(qs, KSTR, kc * 32, lane);

  v8f o[8];
  float mi[8], li[8];
#pragma unroll
  for (int ft = 0; ft < 8; ++ft) { v8f z = {}; o[ft] = z; }
#pragma unroll
  for (int vv = 0; vv < 8; ++vv) { mi[vv] = -3.0e38f; li[vv] = 0.f; }

  int h2 = lane >> 4, nn = lane & 15;
  int kbmax = (q0 + 15) >> 5;                 // inclusive, causal bound
  for (int kb = 0; kb <= kbmax; ++kb) {
    __syncthreads();
    // stage K (key-major) and V (transposed, feat-major), vectorized
#pragma unroll 4
    for (int it = 0; it < 32; ++it) {
      int idx = lane + it * 32;                // 0..1023 float4 chunks
      int r = idx >> 5, c4 = (idx & 31) * 4;
      size_t base = (size_t)(kb * 32 + r) * (cHKV * cDH) + kvh * cDH + c4;
      float4 kv4 = *(const float4*)(k + base);
      uint2 pk = { pack_bf2(kv4.x, kv4.y), pack_bf2(kv4.z, kv4.w) };
      *(uint2*)&ks[r * KSTR + c4] = pk;
      float4 vv4 = *(const float4*)(v + base);
      vs[(c4 + 0) * VSTR + r] = f2bf(vv4.x);
      vs[(c4 + 1) * VSTR + r] = f2bf(vv4.y);
      vs[(c4 + 2) * VSTR + r] = f2bf(vv4.z);
      vs[(c4 + 3) * VSTR + r] = f2bf(vv4.w);
    }
    __syncthreads();

    // S = Q K^T  (two 16x16 score tiles)
    v8f s01[2];
#pragma unroll
    for (int t2 = 0; t2 < 2; ++t2) {
      v8f sacc = {};
#pragma unroll
      for (int kc = 0; kc < 4; ++kc) {
        v16bf bf = frag_b(ks + (t2 * 16) * KSTR, KSTR, kc * 32, lane);
        sacc = wmma_bf16(qa[kc], bf, sacc);
      }
      int ng = kb * 32 + t2 * 16 + nn;        // global key index of this lane
#pragma unroll
      for (int vv = 0; vv < 8; ++vv) {
        int qi = q0 + h2 * 8 + vv;
        float sv = sacc[vv] * scale;
        if (ng > qi) sv = -3.0e38f;           // causal mask
        sacc[vv] = sv;
      }
      s01[t2] = sacc;
    }

    // online softmax: row max / rescale / exp / row sum
    float alpha[8];
#pragma unroll
    for (int vv = 0; vv < 8; ++vv) {
      float mx = fmaxf(s01[0][vv], s01[1][vv]);
#pragma unroll
      for (int off = 8; off > 0; off >>= 1) mx = fmaxf(mx, __shfl_xor(mx, off, 16));
      float mn = fmaxf(mi[vv], mx);
      alpha[vv] = __expf(mi[vv] - mn);
      mi[vv] = mn;
    }
#pragma unroll
    for (int t2 = 0; t2 < 2; ++t2) {
#pragma unroll
      for (int vv = 0; vv < 8; ++vv) {
        float pe = __expf(s01[t2][vv] - mi[vv]);
        s01[t2][vv] = pe;
        ps[(h2 * 8 + vv) * VSTR + t2 * 16 + nn] = f2bf(pe);
      }
    }
#pragma unroll
    for (int vv = 0; vv < 8; ++vv) {
      float su = s01[0][vv] + s01[1][vv];
#pragma unroll
      for (int off = 8; off > 0; off >>= 1) su += __shfl_xor(su, off, 16);
      li[vv] = li[vv] * alpha[vv] + su;
    }
#pragma unroll
    for (int ft = 0; ft < 8; ++ft)
#pragma unroll
      for (int vv = 0; vv < 8; ++vv) o[ft][vv] *= alpha[vv];

    __syncthreads();
    // O += P V  (A = P 16x32, B = V 32x16 per feature tile)
    v16bf pa = frag_a(ps, VSTR, 0, lane);
#pragma unroll
    for (int ft = 0; ft < 8; ++ft) {
      v16bf vb = frag_b(vs + (ft * 16) * VSTR, VSTR, 0, lane);
      o[ft] = wmma_bf16(pa, vb, o[ft]);
    }
  }

  // normalize and write ctx
#pragma unroll
  for (int ft = 0; ft < 8; ++ft)
#pragma unroll
    for (int vv = 0; vv < 8; ++vv) {
      int m = q0 + h2 * 8 + vv;
      ctx[(size_t)m * (cH * cDH) + hh * cDH + ft * 16 + nn] = o[ft][vv] / li[vv];
    }
}

// ----- SwiGLU ---------------------------------------------------------------

__global__ __launch_bounds__(256) void silu_mul_k(float* __restrict__ up,
                                                  const float* __restrict__ gate,
                                                  int n) {
  int i = blockIdx.x * 256 + threadIdx.x;
  if (i < n) {
    float g = gate[i];
    up[i] *= g / (1.f + __expf(-g));
  }
}

// ----- driver ---------------------------------------------------------------

static inline void launch_gemm(const float* A, const float* W, const float* res,
                               float* C, int M, int N, int K, hipStream_t s) {
  dim3 grid((N + GBN - 1) / GBN, (M + GBM - 1) / GBM);
  gemm_bf16_k<<<grid, 256, 0, s>>>(A, W, res, C, M, N, K);
}

extern "C" void kernel_launch(void* const* d_in, const int* in_sizes, int n_in,
                              void* d_out, int out_size, void* d_ws, size_t ws_size,
                              hipStream_t stream) {
  const int*   x_ids     = (const int*)d_in[0];
  const float* tok_embed = (const float*)d_in[1];
  const float* q_w       = (const float*)d_in[2];
  const float* k_w       = (const float*)d_in[3];
  const float* v_w       = (const float*)d_in[4];
  const float* o_w       = (const float*)d_in[5];
  const float* q_norm_w  = (const float*)d_in[6];
  const float* k_norm_w  = (const float*)d_in[7];
  const float* norm1_w   = (const float*)d_in[8];
  const float* norm2_w   = (const float*)d_in[9];
  const float* up_w      = (const float*)d_in[10];
  const float* gate_w    = (const float*)d_in[11];
  const float* down_w    = (const float*)d_in[12];
  const float* final_w   = (const float*)d_in[13];
  float* out = (float*)d_out;

  float* ws  = (float*)d_ws;
  float* X   = ws;                                  // [T][D] residual
  float* Hn  = X   + (size_t)cT * cD;               // [T][D] normed
  float* Q   = Hn  + (size_t)cT * cD;               // [T][H*DH]
  float* Kb  = Q   + (size_t)cT * cH * cDH;         // [T][HKV*DH]
  float* Vb  = Kb  + (size_t)cT * cHKV * cDH;       // [T][HKV*DH]
  float* CTX = Vb  + (size_t)cT * cHKV * cDH;       // [T][H*DH]
  float* UP  = CTX + (size_t)cT * cH * cDH;         // [T][FF]
  float* GT  = UP  + (size_t)cT * cFF;              // [T][FF]

  embed_k<<<(cT * cD) / 256, 256, 0, stream>>>(x_ids, tok_embed, X);

  for (int l = 0; l < cL; ++l) {
    const float* qw = q_w    + (size_t)l * cH * cDH * cD;
    const float* kw = k_w    + (size_t)l * cHKV * cDH * cD;
    const float* vw = v_w    + (size_t)l * cHKV * cDH * cD;
    const float* ow = o_w    + (size_t)l * cD * cH * cDH;
    const float* uw = up_w   + (size_t)l * cFF * cD;
    const float* gw = gate_w + (size_t)l * cFF * cD;
    const float* dw = down_w + (size_t)l * cD * cFF;

    rmsnorm_k<<<cT, 256, 0, stream>>>(X, norm1_w + (size_t)l * cD, Hn);

    launch_gemm(Hn, qw, nullptr, Q,  cT, cH * cDH,   cD, stream);
    launch_gemm(Hn, kw, nullptr, Kb, cT, cHKV * cDH, cD, stream);
    launch_gemm(Hn, vw, nullptr, Vb, cT, cHKV * cDH, cD, stream);

    rope_rms_k<<<cT * cH,   32, 0, stream>>>(Q,  cH,   cH * cDH,
                                             q_norm_w + (size_t)l * cDH);
    rope_rms_k<<<cT * cHKV, 32, 0, stream>>>(Kb, cHKV, cHKV * cDH,
                                             k_norm_w + (size_t)l * cDH);

    attn_k<<<dim3(cH, cT / 16), 32, 0, stream>>>(Q, Kb, Vb, CTX);

    launch_gemm(CTX, ow, X, X, cT, cD, cH * cDH, stream);   // x += ctx @ o^T

    rmsnorm_k<<<cT, 256, 0, stream>>>(X, norm2_w + (size_t)l * cD, Hn);

    launch_gemm(Hn, uw, nullptr, UP, cT, cFF, cD, stream);
    launch_gemm(Hn, gw, nullptr, GT, cT, cFF, cD, stream);

    int nff = cT * cFF;
    silu_mul_k<<<(nff + 255) / 256, 256, 0, stream>>>(UP, GT, nff);

    launch_gemm(UP, dw, X, X, cT, cD, cFF, stream);          // x += mlp
  }

  rmsnorm_k<<<cT, 256, 0, stream>>>(X, final_w, Hn);
  launch_gemm(Hn, tok_embed, nullptr, out, cT, cV, cD, stream);  // logits
}
